// QLinear_75265006895498
// MI455X (gfx1250) — matrix operand order repaired and checked by reading, not compiled, not run on hardware
//
#include <hip/hip_runtime.h>

// ---------- problem constants (from the reference) ----------
#define K_DIM   4096      // IN_F
#define N_DIM   11008     // OUT_F
#define M_DIM   8192      // BATCH * SEQ

// ---------- tiling ----------
#define BM 128
#define BN 128
#define BK 32             // one wmma_f32_16x16x32_f16 K-step
#define LDT 40            // LDS row stride in halves: 32 + 8 pad (80B, 16B aligned)

typedef _Float16 v8h  __attribute__((ext_vector_type(8)));
typedef _Float16 v16h __attribute__((ext_vector_type(16)));
typedef float    v4f  __attribute__((ext_vector_type(4)));
typedef float    v8f  __attribute__((ext_vector_type(8)));
typedef int      v4i  __attribute__((ext_vector_type(4)));

__global__ __launch_bounds__(256)
void qlinear_wmma_kernel(const float* __restrict__ X,   // [M, K] fp32
                         const int*   __restrict__ Wq,  // [N, K] int32 (values 0..15)
                         const float* __restrict__ S,   // [N] fp32
                         float*       __restrict__ Y)   // [M, N] fp32
{
    __shared__ __align__(16) _Float16 Alds[BM * LDT];   // x tile, f16
    __shared__ __align__(16) _Float16 Blds[BN * LDT];   // dequantized W tile, f16

    const int tid   = threadIdx.x;
    const int lane  = tid & 31;
    const int wid   = tid >> 5;      // 0..7
    const int waveM = wid >> 2;      // 0..1 -> 64-row slab
    const int waveN = wid & 3;       // 0..3 -> 32-col slab

    const int n0 = blockIdx.x * BN;  // N-major grid: W tile reused across M blocks via L2
    const int m0 = blockIdx.y * BM;

    // ---- loader mapping: thread t loads 16 elems of row (t/2), half (t&1) ----
    const int lr = tid >> 1;         // 0..127
    const int lh = tid & 1;          // 0/1 -> columns lh*16 .. lh*16+15
    const float sc = S[n0 + lr];     // per-output-channel scale (fixed across K loop)

    const float* xrow = X  + (size_t)(m0 + lr) * K_DIM + lh * 16;
    const int*   wrow = Wq + (size_t)(n0 + lr) * K_DIM + lh * 16;

    const int lo = lane & 15;        // column / row-within-16
    const int hi = lane >> 4;        // K-half selector for fragments

    v8f acc[4][2];
    #pragma unroll
    for (int wm = 0; wm < 4; ++wm)
        #pragma unroll
        for (int wn = 0; wn < 2; ++wn)
            acc[wm][wn] = (v8f){0.f, 0.f, 0.f, 0.f, 0.f, 0.f, 0.f, 0.f};

    for (int k0 = 0; k0 < K_DIM; k0 += BK) {
        // ---- global loads: 4x float4 of x, 4x int4 of Wq per thread ----
        v4f xf[4];
        v4i wi[4];
        #pragma unroll
        for (int i = 0; i < 4; ++i) {
            xf[i] = *(const v4f*)(xrow + k0 + i * 4);
            wi[i] = *(const v4i*)(wrow + k0 + i * 4);
        }

        __syncthreads();   // previous iteration's LDS reads complete

        // ---- convert to f16 (dequantize W) and stage into LDS ----
        v8h a0, a1, b0, b1;
        #pragma unroll
        for (int i = 0; i < 8; ++i) {
            a0[i] = (_Float16)xf[i >> 2][i & 3];
            a1[i] = (_Float16)xf[2 + (i >> 2)][i & 3];
            b0[i] = (_Float16)((float)wi[i >> 2][i & 3] * sc);
            b1[i] = (_Float16)((float)wi[2 + (i >> 2)][i & 3] * sc);
        }
        {
            _Float16* ap = &Alds[lr * LDT + lh * 16];
            _Float16* bp = &Blds[lr * LDT + lh * 16];
            *(v8h*)(ap)     = a0;
            *(v8h*)(ap + 8) = a1;
            *(v8h*)(bp)     = b0;
            *(v8h*)(bp + 8) = b1;
        }

        __syncthreads();   // tile visible to all waves

        // ---- load fragments (ISA 7.12.2 layouts) and issue 8 WMMAs ----
        v16h afrag[4];
        #pragma unroll
        for (int wm = 0; wm < 4; ++wm) {
            const int m = waveM * 64 + wm * 16 + lo;
            v8h t0 = *(const v8h*)&Alds[m * LDT + hi * 8];        // K 0..7  (or 8..15)
            v8h t1 = *(const v8h*)&Alds[m * LDT + 16 + hi * 8];   // K 16..23 (or 24..31)
            afrag[wm] = __builtin_shufflevector(t0, t1,
                0, 1, 2, 3, 4, 5, 6, 7, 8, 9, 10, 11, 12, 13, 14, 15);
        }
        v16h bfrag[2];
        #pragma unroll
        for (int wn = 0; wn < 2; ++wn) {
            const int n = waveN * 32 + wn * 16 + lo;
            v8h t0 = *(const v8h*)&Blds[n * LDT + hi * 8];        // K 0..7  (or 8..15)
            v8h t1 = *(const v8h*)&Blds[n * LDT + 16 + hi * 8];   // K 16..23 (or 24..31)
            bfrag[wn] = __builtin_shufflevector(t0, t1,
                0, 1, 2, 3, 4, 5, 6, 7, 8, 9, 10, 11, 12, 13, 14, 15);
        }

        #pragma unroll
        for (int wm = 0; wm < 4; ++wm)
            #pragma unroll
            for (int wn = 0; wn < 2; ++wn)
                acc[wm][wn] = __builtin_amdgcn_wmma_f32_16x16x32_f16(
                    /*neg_a=*/false, afrag[wm],
                    /*neg_b=*/false, bfrag[wn],
                    /*c_mod=*/(short)0, acc[wm][wn],
                    /*reuse_a=*/false, /*reuse_b=*/false);
    }

    // ---- epilogue: C/D layout -> row = base + hi*8 + i, col = lo (coalesced) ----
    #pragma unroll
    for (int wm = 0; wm < 4; ++wm) {
        const int mbase = m0 + waveM * 64 + wm * 16 + hi * 8;
        #pragma unroll
        for (int wn = 0; wn < 2; ++wn) {
            const int nn = n0 + waveN * 32 + wn * 16 + lo;
            #pragma unroll
            for (int i = 0; i < 8; ++i)
                Y[(size_t)(mbase + i) * N_DIM + nn] = acc[wm][wn][i];
        }
    }
}

extern "C" void kernel_launch(void* const* d_in, const int* in_sizes, int n_in,
                              void* d_out, int out_size, void* d_ws, size_t ws_size,
                              hipStream_t stream) {
    (void)in_sizes; (void)n_in; (void)d_ws; (void)ws_size; (void)out_size;
    const float* x  = (const float*)d_in[0];   // [4, 2048, 4096] fp32
    const int*   wq = (const int*)d_in[1];     // [11008, 4096] int32
    const float* s  = (const float*)d_in[2];   // [11008] fp32
    float*       y  = (float*)d_out;           // [4, 2048, 11008] fp32

    dim3 grid(N_DIM / BN, M_DIM / BM);         // (86, 64)
    dim3 block(256);
    qlinear_wmma_kernel<<<grid, block, 0, stream>>>(x, wq, s, y);
}
